// MultiHeadAttention_14860586844473
// MI455X (gfx1250) — matrix operand order repaired and checked by reading, not compiled, not run on hardware
//
#include <hip/hip_runtime.h>
#include <hip/hip_bf16.h>

// ---------------------------------------------------------------------------
// MHA forward for MI455X (gfx1250, wave32, WMMA).
// All matrix math runs on v_wmma_f32_16x16x32_bf16 (f32 accumulate).
// ---------------------------------------------------------------------------

typedef __bf16 bf16_t;
typedef __attribute__((ext_vector_type(16))) __bf16 v16bf;
typedef __attribute__((ext_vector_type(8)))  __bf16 v8bf;
typedef __attribute__((ext_vector_type(8)))  float  v8f;

union V16u { v16bf v; v8bf h[2]; };

#define BB   2
#define TT   2048
#define DD   1024
#define HH   16
#define HDD  64
#define MM   (BB * TT)          // 4096 rows
#define ATT_SCALE 0.125f        // 1/sqrt(64)

// ---------------------------------------------------------------------------
// fp32 -> bf16 conversion (bandwidth trivial)
// ---------------------------------------------------------------------------
__global__ __launch_bounds__(256) void cvt_f32_bf16(const float* __restrict__ src,
                                                    bf16_t* __restrict__ dst, int n) {
#if defined(__HIP_DEVICE_COMPILE__)
  int i = (blockIdx.x * blockDim.x + threadIdx.x) * 4;
  if (i + 3 < n) {
    float4 f = *(const float4*)(src + i);
    dst[i + 0] = (bf16_t)f.x;
    dst[i + 1] = (bf16_t)f.y;
    dst[i + 2] = (bf16_t)f.z;
    dst[i + 3] = (bf16_t)f.w;
  } else {
    for (; i < n; ++i) dst[i] = (bf16_t)src[i];
  }
#endif
}

// ---------------------------------------------------------------------------
// WMMA GEMM: C[M,N] = A[M,K] * W[N,K]^T + bias  (A, W bf16 row-major)
//   mode 0: outF[m*N + n] = (acc + bias) * scale          (fp32, final proj)
//   mode 1: outH[(b*H+h)*T*HD + t*HD + hd] = bf16(...)    (QKV layout)
// One 16x64 strip per wave (4 N-tiles share one A fragment -> 4 WMMAs per
// A load), 8 waves per block.
// A fragment (16x32 bf16): lane row = lane&15; half-chunks at (lane>>4)*8, +16.
// B fragment (32x16 bf16): lane col = lane&15; half-chunks at (lane>>4)*8, +16.
// C/D layout: reg r -> row r + 8*(lane>>4), col = lane&15.
// ---------------------------------------------------------------------------
__global__ __launch_bounds__(256) void gemm_wmma(const bf16_t* __restrict__ A,
                                                 const bf16_t* __restrict__ W,
                                                 const float*  __restrict__ bias,
                                                 float*        __restrict__ outF,
                                                 bf16_t*       __restrict__ outH,
                                                 int M, int N, int K,
                                                 float scale, int mode) {
#if defined(__HIP_DEVICE_COMPILE__)
  const int w      = threadIdx.x >> 5;
  const int lane   = threadIdx.x & 31;
  const int tilesN = N >> 6;                 // 64-wide supertiles
  const int tile   = blockIdx.x * 8 + w;
  const int m0     = (tile / tilesN) << 4;
  const int n0     = (tile % tilesN) << 6;
  if (m0 >= M) return;

  const int lcol   = lane & 15;
  const int ksel8  = (lane >> 4) << 3;   // A chunk select: 0 or 8
  const int ksel16 = (lane >> 4) << 4;   // unused (B also selects with ksel8): 0 or 16

  const bf16_t* arow = A + (size_t)(m0 + lcol) * K;
  const bf16_t* wrow[4];
#pragma unroll
  for (int j = 0; j < 4; ++j)
    wrow[j] = W + (size_t)(n0 + 16 * j + lcol) * K;

  v8f c[4] = {{}, {}, {}, {}};
  for (int k0 = 0; k0 < K; k0 += 32) {
    // pull the next A chunk toward the caches while WMMAs drain
    __builtin_prefetch(arow + k0 + 256, 0, 1);
    V16u ua;
    ua.h[0] = *(const v8bf*)(arow + k0 + ksel8);
    ua.h[1] = *(const v8bf*)(arow + k0 + ksel8 + 16);
#pragma unroll
    for (int j = 0; j < 4; ++j) {
      V16u ub; ub.h[0] = *(const v8bf*)(wrow[j] + k0 + ksel8); ub.h[1] = *(const v8bf*)(wrow[j] + k0 + ksel8 + 16); v16bf bfrag = ub.v;
      c[j] = __builtin_amdgcn_wmma_f32_16x16x32_bf16(false, ua.v, false, bfrag,
                                                     (short)0, c[j], false, false);
    }
  }

  const int rofs = (lane >> 4) << 3;
#pragma unroll
  for (int j = 0; j < 4; ++j) {
    const int n  = n0 + 16 * j + lcol;
    const float bn = bias[n];
    if (mode == 0) {
#pragma unroll
      for (int r = 0; r < 8; ++r) {
        const int mrow = m0 + r + rofs;
        outF[(size_t)mrow * N + n] = (c[j][r] + bn) * scale;
      }
    } else {
      const int h  = n >> 6;               // / HD
      const int hd = n & (HDD - 1);
#pragma unroll
      for (int r = 0; r < 8; ++r) {
        const int mrow = m0 + r + rofs;
        const int bb = mrow >> 11;         // / T
        const int t  = mrow & (TT - 1);
        outH[(((size_t)(bb * HH + h)) * TT + t) * HDD + hd] =
            (bf16_t)((c[j][r] + bn) * scale);
      }
    }
  }
#endif
}

// ---------------------------------------------------------------------------
// Flash attention: one wave handles a 16-row Q tile for one (b,h).
// Q/K/V are bf16 in [B*H, T, HD]; Q pre-scaled by 1/sqrt(HD).
// Per 32-key step: 4 WMMAs for S = Q*K^T, online softmax, 4 WMMAs for O += P*V.
// P is reshaped C-layout -> A-layout via 1KB/wave LDS; V staged transposed in
// 4KB/wave LDS so PV B-fragments are two v8bf row reads per lane.
// ---------------------------------------------------------------------------
__global__ __launch_bounds__(128) void attn_wmma(const bf16_t* __restrict__ Q,
                                                 const bf16_t* __restrict__ Kmat,
                                                 const bf16_t* __restrict__ Vmat,
                                                 const unsigned char* __restrict__ kpm,
                                                 bf16_t* __restrict__ Ob) {
#if defined(__HIP_DEVICE_COMPILE__)
  __shared__ __align__(32) bf16_t lds_p[4][16 * 32];   // P tile  (16 rows x 32 keys)
  __shared__ __align__(32) bf16_t lds_vt[4][64 * 32];  // V^T     (64 hd   x 32 keys)

  const int w    = threadIdx.x >> 5;
  const int lane = threadIdx.x & 31;
  const int tile = blockIdx.x * 4 + w;     // 4096 tiles total
  const int bh   = tile >> 7;              // / (T/16)
  const int it   = tile & 127;
  const int b    = bh >> 4;                // / H
  const int h    = bh & (HH - 1);
  const int i0   = it << 4;

  const int lcol   = lane & 15;
  const int lrow   = (lane >> 4) << 3;     // C-layout row offset: 0 or 8
  const int ksel8  = (lane >> 4) << 3;
  const int ksel16 = (lane >> 4) << 4;
  const size_t baseBH = (size_t)bh * TT * HDD;

  // Q fragments (A operand), hd chunks [0,32) and [32,64)
  v16bf qa[2];
  {
    const bf16_t* qp = Q + baseBH + (size_t)(i0 + lcol) * HDD;
#pragma unroll
    for (int kf = 0; kf < 2; ++kf) {
      V16u u;
      u.h[0] = *(const v8bf*)(qp + kf * 32 + ksel8);
      u.h[1] = *(const v8bf*)(qp + kf * 32 + ksel8 + 16);
      qa[kf] = u.v;
    }
  }

  v8f o0 = {}, o1 = {}, o2 = {}, o3 = {};
  float mrun[8], lrun[8];
#pragma unroll
  for (int r = 0; r < 8; ++r) { mrun[r] = -1e30f; lrun[r] = 0.0f; }

  const int jmax = i0 + 16;                // causal: keys j <= i0+15
  for (int j0 = 0; j0 < jmax; j0 += 32) {
    // ---- scores S(16x32) = Q(16x64) * K^T(64x32): two 16-col tiles ----
    v8f s0 = {}, s1 = {};
    int jr0 = j0 + lcol;      if (jr0 > TT - 1) jr0 = TT - 1;
    int jr1 = j0 + 16 + lcol; if (jr1 > TT - 1) jr1 = TT - 1;
    const bf16_t* kp0 = Kmat + baseBH + (size_t)jr0 * HDD + ksel8;
    const bf16_t* kp1 = Kmat + baseBH + (size_t)jr1 * HDD + ksel8;
#pragma unroll
    for (int kf = 0; kf < 2; ++kf) {
      V16u uk0, uk1;
      uk0.h[0] = *(const v8bf*)(kp0 + kf * 32); uk0.h[1] = *(const v8bf*)(kp0 + kf * 32 + 16);
      uk1.h[0] = *(const v8bf*)(kp1 + kf * 32); uk1.h[1] = *(const v8bf*)(kp1 + kf * 32 + 16);
      v16bf kb0 = uk0.v;
      v16bf kb1 = uk1.v;
      s0 = __builtin_amdgcn_wmma_f32_16x16x32_bf16(false, qa[kf], false, kb0,
                                                   (short)0, s0, false, false);
      s1 = __builtin_amdgcn_wmma_f32_16x16x32_bf16(false, qa[kf], false, kb1,
                                                   (short)0, s1, false, false);
    }

    // ---- stage V tile transposed into LDS: lds_vt[hd*32 + k] ----
    {
      int jr = j0 + lane; if (jr > TT - 1) jr = TT - 1;
      const bf16_t* vr = Vmat + baseBH + (size_t)jr * HDD;
#pragma unroll
      for (int cch = 0; cch < 8; ++cch) {
        v8bf chunk = *(const v8bf*)(vr + cch * 8);
#pragma unroll
        for (int e = 0; e < 8; ++e)
          lds_vt[w][(cch * 8 + e) * 32 + lane] = chunk[e];
      }
    }

    // ---- mask + online softmax (row r lives in reg r, 16 lanes per row) ----
    const int gj0c = j0 + lcol, gj1c = j0 + 16 + lcol;
    const bool ok0 = (gj0c < TT) && (kpm[(size_t)b * TT + gj0c] != 0);
    const bool ok1 = (gj1c < TT) && (kpm[(size_t)b * TT + gj1c] != 0);
    float fscale[8];
#pragma unroll
    for (int r = 0; r < 8; ++r) {
      const int gi = i0 + r + lrow;
      float e0 = (ok0 && gj0c <= gi) ? s0[r] : -1e30f;
      float e1 = (ok1 && gj1c <= gi) ? s1[r] : -1e30f;
      float vmx = fmaxf(e0, e1);
#pragma unroll
      for (int off = 1; off < 16; off <<= 1)
        vmx = fmaxf(vmx, __shfl_xor(vmx, off, 16));
      const float mnew = fmaxf(mrun[r], vmx);
      const float fac  = __expf(mrun[r] - mnew);
      const float p0 = __expf(e0 - mnew);
      const float p1 = __expf(e1 - mnew);
      float rs = p0 + p1;
#pragma unroll
      for (int off = 1; off < 16; off <<= 1)
        rs += __shfl_xor(rs, off, 16);
      lrun[r] = lrun[r] * fac + rs;
      mrun[r] = mnew;
      fscale[r] = fac;
      const int prow = r + lrow;
      lds_p[w][prow * 32 + lcol]      = (bf16_t)p0;
      lds_p[w][prow * 32 + 16 + lcol] = (bf16_t)p1;
    }
#pragma unroll
    for (int r = 0; r < 8; ++r) {
      o0[r] *= fscale[r]; o1[r] *= fscale[r];
      o2[r] *= fscale[r]; o3[r] *= fscale[r];
    }
    __builtin_amdgcn_wave_barrier();   // order LDS P/V^T writes vs. reads (same wave)

    // ---- P A-fragment from LDS (16x32) ----
    v16bf pa;
    {
      V16u u;
      u.h[0] = *(const v8bf*)(&lds_p[w][lcol * 32 + ksel8]);
      u.h[1] = *(const v8bf*)(&lds_p[w][lcol * 32 + ksel8 + 16]);
      pa = u.v;
    }
    // ---- O(16x64) += P(16x32) * V(32x64): 4 N-tiles ----
    {
      V16u uv[4];
#pragma unroll
      for (int q = 0; q < 4; ++q) {
        uv[q].h[0] = *(const v8bf*)(&lds_vt[w][(q * 16 + lcol) * 32 + ksel8]);
        uv[q].h[1] = *(const v8bf*)(&lds_vt[w][(q * 16 + lcol) * 32 + ksel8 + 16]);
      }
      v16bf vb0 = uv[0].v;
      v16bf vb1 = uv[1].v;
      v16bf vb2 = uv[2].v;
      v16bf vb3 = uv[3].v;
      o0 = __builtin_amdgcn_wmma_f32_16x16x32_bf16(false, pa, false, vb0, (short)0, o0, false, false);
      o1 = __builtin_amdgcn_wmma_f32_16x16x32_bf16(false, pa, false, vb1, (short)0, o1, false, false);
      o2 = __builtin_amdgcn_wmma_f32_16x16x32_bf16(false, pa, false, vb2, (short)0, o2, false, false);
      o3 = __builtin_amdgcn_wmma_f32_16x16x32_bf16(false, pa, false, vb3, (short)0, o3, false, false);
    }
    __builtin_amdgcn_wave_barrier();
  }

  // ---- epilogue: O /= l, write bf16 [B*T, D] row-major for final GEMM ----
#pragma unroll
  for (int r = 0; r < 8; ++r) {
    const float inv = (lrun[r] > 0.0f) ? (1.0f / lrun[r]) : 0.0f;
    const int t = i0 + r + lrow;
    const size_t base = ((size_t)b * TT + t) * DD + (size_t)h * HDD + lcol;
    Ob[base +  0] = (bf16_t)(o0[r] * inv);
    Ob[base + 16] = (bf16_t)(o1[r] * inv);
    Ob[base + 32] = (bf16_t)(o2[r] * inv);
    Ob[base + 48] = (bf16_t)(o3[r] * inv);
  }
#endif
}

// ---------------------------------------------------------------------------
// Host launch
// ---------------------------------------------------------------------------
extern "C" void kernel_launch(void* const* d_in, const int* in_sizes, int n_in,
                              void* d_out, int out_size, void* d_ws, size_t ws_size,
                              hipStream_t stream) {
  const float* x  = (const float*)d_in[0];
  const unsigned char* kpm = (const unsigned char*)d_in[1];   // bool mask
  const float* Wq = (const float*)d_in[2];
  const float* bq = (const float*)d_in[3];
  const float* Wk = (const float*)d_in[4];
  const float* bk = (const float*)d_in[5];
  const float* Wv = (const float*)d_in[6];
  const float* bv = (const float*)d_in[7];
  const float* Wo = (const float*)d_in[8];
  const float* bo = (const float*)d_in[9];
  float* out = (float*)d_out;

  const size_t NX = (size_t)MM * DD;     // 4096*1024 elements
  const size_t NW = (size_t)DD * DD;     // 1024*1024 elements

  bf16_t* ws = (bf16_t*)d_ws;
  bf16_t* xb    = ws;                    // x in bf16
  bf16_t* wqb   = xb  + NX;
  bf16_t* wkb   = wqb + NW;
  bf16_t* wvb   = wkb + NW;
  bf16_t* wob   = wvb + NW;
  bf16_t* qb    = wob + NW;              // [B*H, T, HD]
  bf16_t* kb    = qb  + NX;
  bf16_t* vb    = kb  + NX;
  bf16_t* attnb = vb  + NX;              // [B*T, D] bf16
  // total: 5*NX + 4*NW bf16 = 48 MiB

  // 1) convert inputs to bf16
  {
    int n = (int)NX;
    cvt_f32_bf16<<<(n / 4 + 255) / 256, 256, 0, stream>>>(x, xb, n);
    int nw = (int)NW;
    dim3 g((nw / 4 + 255) / 256);
    cvt_f32_bf16<<<g, 256, 0, stream>>>(Wq, wqb, nw);
    cvt_f32_bf16<<<g, 256, 0, stream>>>(Wk, wkb, nw);
    cvt_f32_bf16<<<g, 256, 0, stream>>>(Wv, wvb, nw);
    cvt_f32_bf16<<<g, 256, 0, stream>>>(Wo, wob, nw);
  }

  // 2) QKV projections (WMMA); Q pre-scaled by 1/sqrt(HD)
  const int tiles  = (MM / 16) * (DD / 64);   // 4096 wave-strips
  const int blocks = tiles / 8;               // 512
  gemm_wmma<<<blocks, 256, 0, stream>>>(xb, wqb, bq, nullptr, qb,
                                        MM, DD, DD, ATT_SCALE, 1);
  gemm_wmma<<<blocks, 256, 0, stream>>>(xb, wkb, bk, nullptr, kb,
                                        MM, DD, DD, 1.0f, 1);
  gemm_wmma<<<blocks, 256, 0, stream>>>(xb, wvb, bv, nullptr, vb,
                                        MM, DD, DD, 1.0f, 1);

  // 3) causal flash attention (WMMA): 4096 row-tiles, 4 waves/block
  attn_wmma<<<(BB * HH * (TT / 16)) / 4, 128, 0, stream>>>(qb, kb, vb, kpm, attnb);

  // 4) output projection -> fp32 d_out
  gemm_wmma<<<blocks, 256, 0, stream>>>(attnb, wob, bo, out, nullptr,
                                        MM, DD, DD, 1.0f, 0);
}
